// BiasedMHA_12584254177663
// MI455X (gfx1250) — compile-verified, matrix operand order checked by reading, not run on hardware
//
#include <hip/hip_runtime.h>
#include <hip/hip_bf16.h>

typedef __bf16 bf16;
typedef __attribute__((ext_vector_type(16))) __bf16 v16bf;
typedef __attribute__((ext_vector_type(8)))  __bf16 v8bf;
typedef __attribute__((ext_vector_type(8)))  float  v8f;

// ---------------------------------------------------------------------------
// Fragment loader for v_wmma_f32_16x16x32_bf16 (A and B share the striping):
// lane&15 selects the row (A: m, B: n); halves 0..7 -> K = k0 + (lane&16?8:0)+i,
// halves 8..15 -> +16.  Two contiguous 16B runs -> two b128 loads.
// ---------------------------------------------------------------------------
__device__ __forceinline__ v16bf load_frag(const bf16* __restrict__ base, int ld,
                                           int row, int k0, int lane) {
    const bf16* p = base + (long)row * ld + k0 + ((lane & 16) ? 8 : 0);
    v8bf lo = *(const v8bf*)p;
    v8bf hi = *(const v8bf*)(p + 16);
    v16bf f;
#pragma unroll
    for (int i = 0; i < 8; ++i) { f[i] = lo[i]; f[i + 8] = hi[i]; }
    return f;
}

__device__ __forceinline__ v8f wmma_bf16(v16bf a, v16bf b, v8f c) {
    return __builtin_amdgcn_wmma_f32_16x16x32_bf16(false, a, false, b,
                                                   (short)0, c, false, false);
}

// ---------------------------------------------------------------------------
// CDNA5 async global->LDS DMA (ASYNCcnt-tracked, bypasses VGPRs).
// Generic pointers to LDS carry the LDS offset in their low 32 bits
// ---------------------------------------------------------------------------
__device__ __forceinline__ void async_ld_b128(const void* gptr, void* lptr) {
    asm volatile("global_load_async_to_lds_b128 %0, %1, off"
                 :: "v"((unsigned)(size_t)lptr),
                    "v"((unsigned long long)(size_t)gptr)
                 : "memory");
}
__device__ __forceinline__ void async_ld_b32(const void* gptr, void* lptr) {
    asm volatile("global_load_async_to_lds_b32 %0, %1, off"
                 :: "v"((unsigned)(size_t)lptr),
                    "v"((unsigned long long)(size_t)gptr)
                 : "memory");
}
__device__ __forceinline__ void wait_async0() {
    asm volatile("s_wait_asynccnt 0x0" ::: "memory");
}

// ---------------------------------------------------------------------------
__global__ void cvt_f32_bf16(const float* __restrict__ src, bf16* __restrict__ dst, long n) {
    long i = (long)blockIdx.x * blockDim.x + threadIdx.x;
    if (i < n) dst[i] = (bf16)src[i];
}

// ---------------------------------------------------------------------------
// out[M,Nc] = (A[M,K] @ W[Nc,K]^T + bias[Nc]) * scale
// 64x64 tile per wave: 4 A-frags x 4 B-frags -> 16 WMMAs per 8 fragment loads.
// mode 0: bf16 (M,Nc) | mode 1: bf16 (B,H,D,N) transposed-per-head | mode 2: f32
// ---------------------------------------------------------------------------
__global__ __launch_bounds__(128) void gemm_bias_kernel(
    const bf16* __restrict__ A, const bf16* __restrict__ W,
    const float* __restrict__ bias, void* __restrict__ out,
    int M, int K, int Nc, float scale, int mode,
    int Bdim, int Ndim, int Hdim, int Ddim)
{
    int lane  = threadIdx.x & 31;
    int wave  = threadIdx.x >> 5;
    int gwave = blockIdx.x * 4 + wave;
    int ctiles = Nc / 64;
    int rbase  = (gwave / ctiles) * 64;
    int cbase  = (gwave % ctiles) * 64;
    if (rbase >= M) return;

    v8f acc[4][4];
#pragma unroll
    for (int i = 0; i < 4; ++i)
#pragma unroll
        for (int j = 0; j < 4; ++j) acc[i][j] = v8f{};

    int rlane = lane & 15;

#pragma unroll 1
    for (int k0 = 0; k0 < K; k0 += 32) {
        v16bf a[4];
#pragma unroll
        for (int i = 0; i < 4; ++i)
            a[i] = load_frag(A, K, rbase + i * 16 + rlane, k0, lane);
#pragma unroll
        for (int j = 0; j < 4; ++j) {
            v16bf b = load_frag(W, K, cbase + j * 16 + rlane, k0, lane);
#pragma unroll
            for (int i = 0; i < 4; ++i)
                acc[i][j] = wmma_bf16(a[i], b, acc[i][j]);
        }
    }

    int rowoff = (lane & 16) ? 8 : 0;
#pragma unroll
    for (int j = 0; j < 4; ++j) {
        int col = cbase + j * 16 + rlane;
        float bv = bias[col];
#pragma unroll
        for (int i = 0; i < 4; ++i) {
#pragma unroll
            for (int r = 0; r < 8; ++r) {
                int row = rbase + i * 16 + r + rowoff;
                float v = (acc[i][j][r] + bv) * scale;
                if (mode == 0) {
                    ((bf16*)out)[(long)row * Nc + col] = (bf16)v;
                } else if (mode == 1) {
                    int bb = row / Ndim, nn = row % Ndim;
                    int hh = col / Ddim, dd = col % Ddim;
                    ((bf16*)out)[(((long)(bb * Hdim + hh)) * Ddim + dd) * Ndim + nn] = (bf16)v;
                } else {
                    ((float*)out)[(long)row * Nc + col] = v;
                }
            }
        }
    }
}

// ---------------------------------------------------------------------------
// Fused flash attention. block = (b, 16-query tile), 16 waves = 16 heads.
// attn_bias (dominant 256MB stream) + attn_mask are DMA'd into double-buffered
// LDS with global_load_async_to_lds_b128/_b32 so chunk c+1 streams in while
// chunk c computes; every byte read exactly once, fully coalesced.
// ---------------------------------------------------------------------------
__global__ __launch_bounds__(512) void attn_kernel(
    const bf16* __restrict__ q, const bf16* __restrict__ k,
    const bf16* __restrict__ vt, const float* __restrict__ bias,
    const int* __restrict__ mask, bf16* __restrict__ attn,
    int Bn, int Nn, int Hn, int Dn)
{
    __shared__ __attribute__((aligned(16))) float sBias[2][16][32][16]; // 64 KB
    __shared__ __attribute__((aligned(16))) int   sMask[2][16][32];     //  4 KB
    __shared__ __attribute__((aligned(16))) bf16  sP[16][16][32];       // 16 KB

    int tid  = threadIdx.x;
    int lane = tid & 31;
    int h    = tid >> 5;          // wave index == head
    int b     = blockIdx.y;
    int qbase = blockIdx.x * 16;
    int F = Hn * Dn;

    const bf16* qb = q  + ((long)(b * Nn + qbase)) * F + h * Dn;
    const bf16* kb = k  + ((long)b * Nn) * F + h * Dn;
    const bf16* vb = vt + (((long)(b * Hn + h)) * Dn) * Nn;

    // bias stage decomposition: per chunk 16q x (32k x 16h = 512 floats/row).
    // 128 threads x float4 per row, 4 rows per pass, 4 passes.
    int srow  = tid >> 7;              // 0..3  (row subgroup)
    int scol  = tid & 127;             // float4 index within row
    int skk   = scol >> 2;             // key 0..31
    int shh   = (scol & 3) * 4;        // head 0,4,8,12
    const float* biasRow0 = bias + (((long)b * Nn + qbase) * Nn) * Hn;

    v16bf qf = load_frag(qb, F, lane & 15, 0, lane);   // full D=32

    v8f o0 = {}, o1 = {};
    float rowm[8], rowl[8];
#pragma unroll
    for (int r = 0; r < 8; ++r) { rowm[r] = -1e30f; rowl[r] = 0.f; }

    int rowoff = (lane & 16) ? 8 : 0;
    int keycol = lane & 15;
    int mq = tid >> 5, mk = tid & 31;  // mask stage coords

    // ---- stage chunk 0 into buffer 0
    {
        int kc = 0;
#pragma unroll
        for (int it = 0; it < 4; ++it) {
            int qq = it * 4 + srow;
            async_ld_b128(biasRow0 + ((long)qq * Nn + kc) * Hn + (long)scol * 4,
                          &sBias[0][qq][skk][shh]);
        }
        async_ld_b32(mask + ((long)b * Nn + qbase + mq) * Nn + kc + mk,
                     &sMask[0][mq][mk]);
    }

    int buf = 0;
    int nChunks = Nn / 32;
    for (int c = 0; c < nChunks; ++c) {
        int kc = c * 32;
        wait_async0();        // this wave's DMA for chunk c landed
        __syncthreads();      // everyone's landed; prev compute readers done

        if (c + 1 < nChunks) {       // prefetch chunk c+1 into other buffer
            int kn = kc + 32;
#pragma unroll
            for (int it = 0; it < 4; ++it) {
                int qq = it * 4 + srow;
                async_ld_b128(biasRow0 + ((long)qq * Nn + kn) * Hn + (long)scol * 4,
                              &sBias[buf ^ 1][qq][skk][shh]);
            }
            async_ld_b32(mask + ((long)b * Nn + qbase + mq) * Nn + kn + mk,
                         &sMask[buf ^ 1][mq][mk]);
        }

        // ---- per-head S tile via WMMA
        v16bf kf0 = load_frag(kb + (long)kc * F, F, keycol, 0, lane);
        v16bf kf1 = load_frag(kb + (long)(kc + 16) * F, F, keycol, 0, lane);
        v8f s0 = {}, s1 = {};
        s0 = wmma_bf16(qf, kf0, s0);
        s1 = wmma_bf16(qf, kf1, s1);

#pragma unroll
        for (int r = 0; r < 8; ++r) {
            int ql = r + rowoff;
            float bv0 = sBias[buf][ql][keycol][h];
            float bv1 = sBias[buf][ql][keycol + 16][h];
            int   mk0 = sMask[buf][ql][keycol];
            int   mk1 = sMask[buf][ql][keycol + 16];
            float t0 = mk0 ? -1e30f : s0[r] + bv0;   // mask!=0 -> -inf
            float t1 = mk1 ? -1e30f : s1[r] + bv1;

            float rmax = fmaxf(t0, t1);
#pragma unroll
            for (int off = 1; off < 16; off <<= 1)
                rmax = fmaxf(rmax, __shfl_xor(rmax, off, 32));
            float newm  = fmaxf(rowm[r], rmax);
            float alpha = __expf(rowm[r] - newm);
            float p0 = __expf(t0 - newm);
            float p1 = __expf(t1 - newm);
            float rsum = p0 + p1;
#pragma unroll
            for (int off = 1; off < 16; off <<= 1)
                rsum += __shfl_xor(rsum, off, 32);
            rowl[r] = rowl[r] * alpha + rsum;
            rowm[r] = newm;
            o0[r] *= alpha;
            o1[r] *= alpha;

            sP[h][ql][keycol]      = (bf16)p0;
            sP[h][ql][16 + keycol] = (bf16)p1;
        }
        // sP is wave-private; in-wave DS RAW ordering handled by hw waits

        v16bf pf  = load_frag(&sP[h][0][0], 32, lane & 15, 0, lane);
        v16bf vf0 = load_frag(vb,           Nn, keycol, kc, lane);  // d 0..15
        v16bf vf1 = load_frag(vb + 16 * Nn, Nn, keycol, kc, lane);  // d 16..31
        o0 = wmma_bf16(pf, vf0, o0);
        o1 = wmma_bf16(pf, vf1, o1);

        buf ^= 1;
    }

#pragma unroll
    for (int r = 0; r < 8; ++r) {
        float inv = 1.0f / rowl[r];
        int qr = qbase + r + rowoff;
        bf16* dst = attn + ((long)(b * Nn + qr)) * F + h * Dn;
        dst[keycol]      = (bf16)(o0[r] * inv);
        dst[16 + keycol] = (bf16)(o1[r] * inv);
    }
}

// ---------------------------------------------------------------------------
extern "C" void kernel_launch(void* const* d_in, const int* in_sizes, int n_in,
                              void* d_out, int out_size, void* d_ws, size_t ws_size,
                              hipStream_t stream) {
    const int Bb = 16, Nn = 512, Ff = 512, Hh = 16, Dd = 32;
    const long MN = (long)Bb * Nn;                 // 8192 rows
    const float QSCALE = 5.656854249492380f;       // 1/SCALING = sqrt(D)

    const float* nfeat     = (const float*)d_in[0];
    const float* attn_bias = (const float*)d_in[1];
    const int*   attn_mask = (const int*)d_in[2];
    const float* Wq = (const float*)d_in[3]; const float* bq = (const float*)d_in[4];
    const float* Wk = (const float*)d_in[5]; const float* bk = (const float*)d_in[6];
    const float* Wv = (const float*)d_in[7]; const float* bv = (const float*)d_in[8];
    const float* Wo = (const float*)d_in[9]; const float* bo = (const float*)d_in[10];
    float* out = (float*)d_out;

    bf16* p = (bf16*)d_ws;
    bf16* nfeat_bf = p; p += MN * Ff;
    bf16* Wq_bf = p; p += (long)Ff * Ff;
    bf16* Wk_bf = p; p += (long)Ff * Ff;
    bf16* Wv_bf = p; p += (long)Ff * Ff;
    bf16* Wo_bf = p; p += (long)Ff * Ff;
    bf16* q_bf  = p; p += MN * Ff;
    bf16* k_bf  = p; p += MN * Ff;
    bf16* vt_bf = p; p += MN * Ff;      // (B,H,D,N)
    bf16* at_bf = p; p += MN * Ff;      // attn (B,N,F)

    long nElemA = MN * Ff, nElemW = (long)Ff * Ff;
    cvt_f32_bf16<<<(nElemA + 255) / 256, 256, 0, stream>>>(nfeat, nfeat_bf, nElemA);
    cvt_f32_bf16<<<(nElemW + 255) / 256, 256, 0, stream>>>(Wq, Wq_bf, nElemW);
    cvt_f32_bf16<<<(nElemW + 255) / 256, 256, 0, stream>>>(Wk, Wk_bf, nElemW);
    cvt_f32_bf16<<<(nElemW + 255) / 256, 256, 0, stream>>>(Wv, Wv_bf, nElemW);
    cvt_f32_bf16<<<(nElemW + 255) / 256, 256, 0, stream>>>(Wo, Wo_bf, nElemW);

    // 64x64 tile per wave, 4 waves per block
    int gemmBlocks = (int)(((MN / 64) * (Ff / 64)) / 4);
    gemm_bias_kernel<<<gemmBlocks, 128, 0, stream>>>(
        nfeat_bf, Wq_bf, bq, q_bf, (int)MN, Ff, Ff, QSCALE, 0, Bb, Nn, Hh, Dd);
    gemm_bias_kernel<<<gemmBlocks, 128, 0, stream>>>(
        nfeat_bf, Wk_bf, bk, k_bf, (int)MN, Ff, Ff, 1.0f, 0, Bb, Nn, Hh, Dd);
    gemm_bias_kernel<<<gemmBlocks, 128, 0, stream>>>(
        nfeat_bf, Wv_bf, bv, vt_bf, (int)MN, Ff, Ff, 1.0f, 1, Bb, Nn, Hh, Dd);

    attn_kernel<<<dim3(Nn / 16, Bb), 512, 0, stream>>>(
        q_bf, k_bf, vt_bf, attn_bias, attn_mask, at_bf, Bb, Nn, Hh, Dd);

    gemm_bias_kernel<<<gemmBlocks, 128, 0, stream>>>(
        at_bf, Wo_bf, bo, out, (int)MN, Ff, Ff, 1.0f, 2, Bb, Nn, Hh, Dd);
}
